// RNN_23699629540091
// MI455X (gfx1250) — matrix operand order repaired and checked by reading, not compiled, not run on hardware
//
#include <hip/hip_runtime.h>
#include <math.h>

// Problem constants (match reference)
#define B_   256
#define T_   128
#define V_   348
#define VP_  352     // V padded to multiple of 16 for WMMA tiling
#define E_   512
#define H_   1024
#define G_   3072    // 3*H

typedef __attribute__((ext_vector_type(16))) __bf16 v16bf;
typedef __attribute__((ext_vector_type(8)))  __bf16 v8bf;
typedef __attribute__((ext_vector_type(8)))  float  v8f;

// ---------------------------------------------------------------- utilities
__global__ void zero_u32(unsigned int* __restrict__ p, int n) {
  int i = blockIdx.x * blockDim.x + threadIdx.x;
  if (i < n) p[i] = 0u;
}

__global__ void cvt_f32_bf16(const float* __restrict__ s, __bf16* __restrict__ d, int n) {
  int i = blockIdx.x * blockDim.x + threadIdx.x;
  if (i < n) d[i] = (__bf16)s[i];
}

// w_out [V,H] -> bf16 [VP,H], zero-padded rows 348..351
__global__ void cvt_wout(const float* __restrict__ s, __bf16* __restrict__ d) {
  int i = blockIdx.x * blockDim.x + threadIdx.x;
  if (i >= VP_ * H_) return;
  int row = i >> 10;            // /H_
  int col = i & (H_ - 1);
  d[i] = (row < V_) ? (__bf16)s[row * H_ + col] : (__bf16)0.0f;
}

// e_bf[b, :] = bf16(emb[tok_b(t)])   tok = (t==0) ? bos : target[b, t-1]
__global__ void embed_step(const int* __restrict__ target, const int* __restrict__ bosp,
                           const float* __restrict__ emb, __bf16* __restrict__ ebf, int t) {
  int i = blockIdx.x * blockDim.x + threadIdx.x;       // over B*E (exact)
  int b = i >> 9;                                      // /E_
  int j = i & (E_ - 1);
  int tok = (t == 0) ? bosp[0] : target[b * T_ + (t - 1)];
  ebf[i] = (__bf16)emb[(size_t)tok * E_ + j];
}

// ------------------------------------------------- WMMA bf16 GEMM core
// OUT[16, 16*NT] strip = X_tile[16,K] * W^T. One wave per strip.
// X: [256, K] bf16 row-major.  W: [N, K] bf16 row-major (B-matrix = W^T).
// K compile-time -> immediate-offset loads; explicit 2-stage software
// pipeline so next iteration's loads are in flight during current WMMAs.
// Fragment layouts used here (contiguous k, same order in A and B):
//  A 16x32: lanes 0-15 -> M=lane, K = k0..k0+15 (contiguous)
//           lanes 16-31 -> M=lane-16, K = k0+16..k0+31
//  B 32x16: lanes 0-15 -> N=lane,    K = k0..k0+15 (contiguous)
//           lanes 16-31 -> N=lane-16, K = k0+16..k0+31
//  C/D 16x16 f32: VGPR r -> (M = m0+r | m0+8+r), N = ntile + (lane&15)
template <int NT, int K>
__device__ __forceinline__
void gemm_strip(const __bf16* __restrict__ X, const __bf16* __restrict__ W,
                float* __restrict__ OUT, int ldo, int m0, int n0) {
  const int  lane = threadIdx.x & 31;
  const bool hi   = lane >= 16;

  // single base pointer per operand; all further addressing is constant
  const __bf16* a0 = X + (size_t)(m0 + (lane & 15)) * K + (hi ? 16 : 0);
  const __bf16* b0 = W + (size_t)(n0 + (lane & 15)) * K + (hi ? 16 : 0);

  v8f c[NT];
#pragma unroll
  for (int j = 0; j < NT; ++j) c[j] = (v8f){};

  struct Frag { union { v16bf v; v8bf h[2]; } a; v16bf b[NT]; };
  Frag f[2];

  auto load_frag = [&](Frag& fr, int k0) {
    fr.a.h[0] = *(const v8bf*)(a0 + k0);           // 16B chunk
    fr.a.h[1] = *(const v8bf*)(a0 + k0 + 8);
#pragma unroll
    for (int j = 0; j < NT; ++j)
      fr.b[j] = *(const v16bf*)(b0 + j * 16 * K + k0);   // const offset per j
    __builtin_prefetch((const void*)(b0 + k0 + 512), 0, 1);
  };

  constexpr int NI = K / 32;
  load_frag(f[0], 0);
#pragma unroll
  for (int it = 0; it < NI; ++it) {
    // issue next iteration's loads before consuming current fragments
    if (it + 1 < NI) load_frag(f[(it + 1) & 1], (it + 1) * 32);
    const Frag& fc = f[it & 1];
#pragma unroll
    for (int j = 0; j < NT; ++j)
      c[j] = __builtin_amdgcn_wmma_f32_16x16x32_bf16(
                 false, fc.a.v, false, fc.b[j], (short)0, c[j], false, false);
  }

#pragma unroll
  for (int j = 0; j < NT; ++j) {
    float* orow = OUT + (size_t)(m0 + (hi ? 8 : 0)) * ldo + n0 + j * 16 + (lane & 15);
#pragma unroll
    for (int r = 0; r < 8; ++r) orow[(size_t)r * ldo] = c[j][r];
  }
}

// Fused pair of independent GEMMs (gi and gh of one layer) selected by blockIdx.z:
// doubles resident parallelism vs two serialized same-stream launches.
// amdgpu_waves_per_eu(1): let the RA keep both pipeline stages in registers
// instead of collapsing the B double-buffer to chase occupancy.
template <int NT, int K0, int K1>
__global__ __launch_bounds__(32)
__attribute__((amdgpu_waves_per_eu(1)))
void gemm2_bf16_wmma(const __bf16* __restrict__ X0, const __bf16* __restrict__ W0,
                     float* __restrict__ O0,
                     const __bf16* __restrict__ X1, const __bf16* __restrict__ W1,
                     float* __restrict__ O1, int ldo) {
  const int m0 = blockIdx.x << 4;
  const int n0 = blockIdx.y * (16 * NT);
  if (blockIdx.z == 0) gemm_strip<NT, K0>(X0, W0, O0, ldo, m0, n0);
  else                 gemm_strip<NT, K1>(X1, W1, O1, ldo, m0, n0);
}

template <int NT, int K>
__global__ __launch_bounds__(32)
__attribute__((amdgpu_waves_per_eu(1)))
void gemm_bf16_wmma(const __bf16* __restrict__ X, const __bf16* __restrict__ W,
                    float* __restrict__ OUT, int ldo) {
  gemm_strip<NT, K>(X, W, OUT, ldo, blockIdx.x << 4, blockIdx.y * (16 * NT));
}

// ------------------------------------------------- GRU gate nonlinearity
// gi = x@W_ih^T, gh = h@W_hh^T (no bias yet); PyTorch gate order [r,z,n].
__global__ void gru_update(const float* __restrict__ gi, const float* __restrict__ gh,
                           const float* __restrict__ bih, const float* __restrict__ bhh,
                           float* __restrict__ h, __bf16* __restrict__ hb) {
  int i = blockIdx.x * blockDim.x + threadIdx.x;   // over B*H (exact)
  int b = i >> 10;                                 // /H_
  int j = i & (H_ - 1);
  const float* gib = gi + (size_t)b * G_;
  const float* ghb = gh + (size_t)b * G_;
  float ir = gib[j]            + bih[j];
  float iz = gib[H_ + j]       + bih[H_ + j];
  float in_= gib[2 * H_ + j]   + bih[2 * H_ + j];
  float hr = ghb[j]            + bhh[j];
  float hz = ghb[H_ + j]       + bhh[H_ + j];
  float hn = ghb[2 * H_ + j]   + bhh[2 * H_ + j];
  float r  = 1.0f / (1.0f + expf(-(ir + hr)));
  float z  = 1.0f / (1.0f + expf(-(iz + hz)));
  float n  = tanhf(in_ + r * hn);
  float ho = (1.0f - z) * n + z * h[i];
  h[i]  = ho;
  hb[i] = (__bf16)ho;
}

// ------------------------------------------------- log-softmax + NLL gather
// one block (128 threads) per batch row; LDS tree reductions.
__global__ void logsm_acc(const float* __restrict__ logits, const float* __restrict__ bout,
                          const int* __restrict__ target, float* __restrict__ lp, int t) {
  __shared__ float red[128];
  const int b   = blockIdx.x;
  const int tid = threadIdx.x;
  const float* row = logits + (size_t)b * VP_;

  float mx = -1e30f;
  for (int v = tid; v < V_; v += 128) mx = fmaxf(mx, row[v] + bout[v]);
  red[tid] = mx; __syncthreads();
  for (int s = 64; s > 0; s >>= 1) {
    if (tid < s) red[tid] = fmaxf(red[tid], red[tid + s]);
    __syncthreads();
  }
  mx = red[0]; __syncthreads();

  float sum = 0.0f;
  for (int v = tid; v < V_; v += 128) sum += expf(row[v] + bout[v] - mx);
  red[tid] = sum; __syncthreads();
  for (int s = 64; s > 0; s >>= 1) {
    if (tid < s) red[tid] += red[tid + s];
    __syncthreads();
  }
  if (tid == 0) {
    int tgt = target[b * T_ + t];
    lp[b] += (row[tgt] + bout[tgt]) - mx - logf(red[0]);
  }
}

// ---------------------------------------------------------------- host driver
extern "C" void kernel_launch(void* const* d_in, const int* in_sizes, int n_in,
                              void* d_out, int out_size, void* d_ws, size_t ws_size,
                              hipStream_t stream) {
  (void)in_sizes; (void)n_in; (void)out_size; (void)ws_size;

  const int*   target = (const int*)  d_in[0];
  const int*   bosp   = (const int*)  d_in[1];
  const float* emb    = (const float*)d_in[2];
  const float* w_ih1  = (const float*)d_in[3];
  const float* w_hh1  = (const float*)d_in[4];
  const float* b_ih1  = (const float*)d_in[5];
  const float* b_hh1  = (const float*)d_in[6];
  const float* w_ih2  = (const float*)d_in[7];
  const float* w_hh2  = (const float*)d_in[8];
  const float* b_ih2  = (const float*)d_in[9];
  const float* b_hh2  = (const float*)d_in[10];
  const float* w_ih3  = (const float*)d_in[11];
  const float* w_hh3  = (const float*)d_in[12];
  const float* b_ih3  = (const float*)d_in[13];
  const float* b_hh3  = (const float*)d_in[14];
  const float* w_out  = (const float*)d_in[15];
  const float* b_out  = (const float*)d_in[16];
  float* lp = (float*)d_out;

  // ---- workspace carving (256B aligned slabs)
  char* ws = (char*)d_ws;
  size_t off = 0;
  auto carve = [&](size_t bytes) -> char* {
    char* p = ws + off;
    off += (bytes + 255) & ~(size_t)255;
    return p;
  };
  __bf16* wih1b = (__bf16*)carve((size_t)G_ * E_ * 2);
  __bf16* whh1b = (__bf16*)carve((size_t)G_ * H_ * 2);
  __bf16* wih2b = (__bf16*)carve((size_t)G_ * H_ * 2);
  __bf16* whh2b = (__bf16*)carve((size_t)G_ * H_ * 2);
  __bf16* wih3b = (__bf16*)carve((size_t)G_ * H_ * 2);
  __bf16* whh3b = (__bf16*)carve((size_t)G_ * H_ * 2);
  __bf16* woutb = (__bf16*)carve((size_t)VP_ * H_ * 2);
  __bf16* ebf   = (__bf16*)carve((size_t)B_ * E_ * 2);
  __bf16* h1b   = (__bf16*)carve((size_t)B_ * H_ * 2);
  __bf16* h2b   = (__bf16*)carve((size_t)B_ * H_ * 2);
  __bf16* h3b   = (__bf16*)carve((size_t)B_ * H_ * 2);
  float*  gi    = (float*) carve((size_t)B_ * G_ * 4);
  float*  gh    = (float*) carve((size_t)B_ * G_ * 4);
  float*  h1    = (float*) carve((size_t)B_ * H_ * 4);
  float*  h2    = (float*) carve((size_t)B_ * H_ * 4);
  float*  h3    = (float*) carve((size_t)B_ * H_ * 4);
  float*  logit = (float*) carve((size_t)B_ * VP_ * 4);

  const int TPB = 256;
  auto nb = [](int n, int t) { return (n + t - 1) / t; };

  // ---- one-time-per-call prep: weights fp32 -> bf16 (L2-resident working set)
  cvt_f32_bf16<<<nb(G_ * E_, TPB), TPB, 0, stream>>>(w_ih1, wih1b, G_ * E_);
  cvt_f32_bf16<<<nb(G_ * H_, TPB), TPB, 0, stream>>>(w_hh1, whh1b, G_ * H_);
  cvt_f32_bf16<<<nb(G_ * H_, TPB), TPB, 0, stream>>>(w_ih2, wih2b, G_ * H_);
  cvt_f32_bf16<<<nb(G_ * H_, TPB), TPB, 0, stream>>>(w_hh2, whh2b, G_ * H_);
  cvt_f32_bf16<<<nb(G_ * H_, TPB), TPB, 0, stream>>>(w_ih3, wih3b, G_ * H_);
  cvt_f32_bf16<<<nb(G_ * H_, TPB), TPB, 0, stream>>>(w_hh3, whh3b, G_ * H_);
  cvt_wout<<<nb(VP_ * H_, TPB), TPB, 0, stream>>>(w_out, woutb);

  // ---- state init: h = 0 (fp32 + bf16 mirrors), lp = 0
  zero_u32<<<nb(B_ * H_, TPB), TPB, 0, stream>>>((unsigned int*)h1, B_ * H_);
  zero_u32<<<nb(B_ * H_, TPB), TPB, 0, stream>>>((unsigned int*)h2, B_ * H_);
  zero_u32<<<nb(B_ * H_, TPB), TPB, 0, stream>>>((unsigned int*)h3, B_ * H_);
  zero_u32<<<nb(B_ * H_ / 2, TPB), TPB, 0, stream>>>((unsigned int*)h1b, B_ * H_ / 2);
  zero_u32<<<nb(B_ * H_ / 2, TPB), TPB, 0, stream>>>((unsigned int*)h2b, B_ * H_ / 2);
  zero_u32<<<nb(B_ * H_ / 2, TPB), TPB, 0, stream>>>((unsigned int*)h3b, B_ * H_ / 2);
  zero_u32<<<nb(B_, TPB), TPB, 0, stream>>>((unsigned int*)lp, B_);

  const dim3 gridG2(B_ / 16, G_ / 64, 2);   // 16 x 48 strips x {gi, gh}
  const dim3 gridV(B_ / 16, VP_ / 32);      // 16 x 11 strips (NT=2)
  const dim3 waveBlk(32);

  for (int t = 0; t < T_; ++t) {
    // layer 1: x = emb[tok]
    embed_step<<<nb(B_ * E_, TPB), TPB, 0, stream>>>(target, bosp, emb, ebf, t);
    gemm2_bf16_wmma<4, E_, H_><<<gridG2, waveBlk, 0, stream>>>(ebf, wih1b, gi,
                                                               h1b, whh1b, gh, G_);
    gru_update<<<nb(B_ * H_, TPB), TPB, 0, stream>>>(gi, gh, b_ih1, b_hh1, h1, h1b);
    // layer 2
    gemm2_bf16_wmma<4, H_, H_><<<gridG2, waveBlk, 0, stream>>>(h1b, wih2b, gi,
                                                               h2b, whh2b, gh, G_);
    gru_update<<<nb(B_ * H_, TPB), TPB, 0, stream>>>(gi, gh, b_ih2, b_hh2, h2, h2b);
    // layer 3
    gemm2_bf16_wmma<4, H_, H_><<<gridG2, waveBlk, 0, stream>>>(h2b, wih3b, gi,
                                                               h3b, whh3b, gh, G_);
    gru_update<<<nb(B_ * H_, TPB), TPB, 0, stream>>>(gi, gh, b_ih3, b_hh3, h3, h3b);
    // logits + log-softmax gather
    gemm_bf16_wmma<2, H_><<<gridV, waveBlk, 0, stream>>>(h3b, woutb, logit, VP_);
    logsm_acc<<<B_, 128, 0, stream>>>(logit, b_out, target, lp, t);
  }
}
